// GraphAttentionNetwork_70085276336668
// MI455X (gfx1250) — compile-verified, matrix operand order checked by reading, not run on hardware
//
#include <hip/hip_runtime.h>
#include <hip/hip_bf16.h>

#define NN 50000
#define EE 400000
#define DD 128
#define HH 4
#define LL 3
#define ET (EE + NN)          // edges + self loops = 450000
#define NEG_SLOPE 0.2f
#define LN_EPS 1e-5f

typedef __attribute__((ext_vector_type(16))) __bf16 v16bf;
typedef __attribute__((ext_vector_type(8)))  __bf16 v8bf;
typedef __attribute__((ext_vector_type(8)))  float  v8f;

// ---- ordered-uint encoding of float for atomicMax over signed floats ----
static __device__ __forceinline__ unsigned f2ord(float f) {
  unsigned u = __float_as_uint(f);
  return (u & 0x80000000u) ? ~u : (u | 0x80000000u);
}
static __device__ __forceinline__ float ord2f(unsigned u) {
  unsigned v = (u & 0x80000000u) ? (u & 0x7fffffffu) : ~u;
  return __uint_as_float(v);
}

// =================== prep kernels ===================
__global__ void k_f32_to_bf16(const float* __restrict__ in, __bf16* __restrict__ out, int n) {
  int i = blockIdx.x * blockDim.x + threadIdx.x;
  if (i < n) out[i] = (__bf16)in[i];
}

// W[l] is row-major [128, 512]; produce wt [512, 128] bf16 (column-major of W)
__global__ void k_wt_bf16(const float* __restrict__ W, __bf16* __restrict__ wt) {
  int i = blockIdx.x * blockDim.x + threadIdx.x;
  if (i >= (HH * DD) * DD) return;
  int col = i >> 7;          // 0..511 output column
  int k   = i & 127;         // 0..127
  wt[i] = (__bf16)W[k * (HH * DD) + col];
}

// =================== projection GEMM via WMMA ===================
// h[N,512] = xb[N,128] @ W[128,512]; per wave: 16(M) x 128(N) tile, K in 4 steps of 32.
__global__ __launch_bounds__(128) void k_gemm_wmma(const __bf16* __restrict__ xb,
                                                   const __bf16* __restrict__ wt,   // [512,128]
                                                   float* __restrict__ hbuf) {
  const int wave = threadIdx.x >> 5;      // 0..3 -> N chunk of 128
  const int lane = threadIdx.x & 31;
  const int mbase = blockIdx.x * 16;      // 3125 tiles exactly
  const int nbase = wave * 128;

  const int r    = lane & 15;
  const int koff = (lane < 16) ? 0 : 8;   // 16-bit A layout: chunks [koff,koff+8),[koff+16,koff+24)

  v8f zero = {};
  v8f acc[8];
#pragma unroll
  for (int t = 0; t < 8; ++t) acc[t] = zero;

  const __bf16* arow = xb + (size_t)(mbase + r) * DD;

#pragma unroll
  for (int kk = 0; kk < DD; kk += 32) {
    union { v16bf v; v8bf h[2]; } ua;
    ua.h[0] = *(const v8bf*)(arow + kk + koff);
    ua.h[1] = *(const v8bf*)(arow + kk + koff + 16);
#pragma unroll
    for (int t = 0; t < 8; ++t) {
      const __bf16* brow = wt + (size_t)(nbase + t * 16 + r) * DD + kk + koff;
      union { v16bf v; v8bf h[2]; } ub;
      ub.h[0] = *(const v8bf*)(brow);
      ub.h[1] = *(const v8bf*)(brow + 16);
      acc[t] = __builtin_amdgcn_wmma_f32_16x16x32_bf16(
          /*neg_a=*/false, ua.v, /*neg_b=*/false, ub.v,
          /*c_mod=*/(short)0, acc[t], /*reuse_a=*/false, /*reuse_b=*/false);
    }
  }

  // C layout: VGPR v -> row (lane<16 ? v : v+8), col = lane&15
  const int rowbase = mbase + ((lane < 16) ? 0 : 8);
#pragma unroll
  for (int t = 0; t < 8; ++t) {
    float* out = hbuf + (size_t)rowbase * (HH * DD) + nbase + t * 16 + r;
#pragma unroll
    for (int v = 0; v < 8; ++v) out[(size_t)v * (HH * DD)] = acc[t][v];
  }
}

// =================== per-node attention logits ===================
// one wave per (node, head): al_src = <h, a_src>, al_dst = <h, a_dst>
__global__ void k_attn_logits(const float* __restrict__ hbuf,
                              const float* __restrict__ a_src,
                              const float* __restrict__ a_dst,
                              float* __restrict__ alS, float* __restrict__ alD) {
  int gw   = (blockIdx.x * blockDim.x + threadIdx.x) >> 5;
  int lane = threadIdx.x & 31;
  if (gw >= NN * HH) return;
  int n = gw >> 2, h = gw & 3;
  float4 x = ((const float4*)(hbuf + (size_t)n * (HH * DD) + h * DD))[lane];
  float4 s = ((const float4*)(a_src + h * DD))[lane];
  float4 d = ((const float4*)(a_dst + h * DD))[lane];
  float ps = x.x * s.x + x.y * s.y + x.z * s.z + x.w * s.w;
  float pd = x.x * d.x + x.y * d.y + x.z * d.z + x.w * d.w;
#pragma unroll
  for (int off = 16; off; off >>= 1) {
    ps += __shfl_down(ps, off, 32);
    pd += __shfl_down(pd, off, 32);
  }
  if (lane == 0) { alS[gw] = ps; alD[gw] = pd; }
}

// =================== edge logits + segment max (float4 path) ===================
__global__ void k_edge_logits(const int* __restrict__ srcI, const int* __restrict__ dstI,
                              const float* __restrict__ alS, const float* __restrict__ alD,
                              float* __restrict__ ebuf, unsigned* __restrict__ mord) {
  int e = blockIdx.x * blockDim.x + threadIdx.x;
  if (e >= ET) return;
  int s = (e < EE) ? srcI[e] : (e - EE);
  int d = (e < EE) ? dstI[e] : (e - EE);
  float4 vs = ((const float4*)alS)[s];
  float4 vd = ((const float4*)alD)[d];
  float4 v;
  v.x = vs.x + vd.x; v.y = vs.y + vd.y; v.z = vs.z + vd.z; v.w = vs.w + vd.w;
  v.x = (v.x > 0.f) ? v.x : v.x * NEG_SLOPE;
  v.y = (v.y > 0.f) ? v.y : v.y * NEG_SLOPE;
  v.z = (v.z > 0.f) ? v.z : v.z * NEG_SLOPE;
  v.w = (v.w > 0.f) ? v.w : v.w * NEG_SLOPE;
  ((float4*)ebuf)[e] = v;
  unsigned* mrow = mord + d * HH;
  atomicMax(mrow + 0, f2ord(v.x));
  atomicMax(mrow + 1, f2ord(v.y));
  atomicMax(mrow + 2, f2ord(v.z));
  atomicMax(mrow + 3, f2ord(v.w));
}

// =================== exp(e - max) + segment sum (float4 path) ===================
__global__ void k_exp_den(const int* __restrict__ dstI, const unsigned* __restrict__ mord,
                          float* __restrict__ ebuf, float* __restrict__ den) {
  int e = blockIdx.x * blockDim.x + threadIdx.x;
  if (e >= ET) return;
  int d = (e < EE) ? dstI[e] : (e - EE);
  uint4 m = ((const uint4*)mord)[d];
  float4 v = ((const float4*)ebuf)[e];
  float4 ex;
  ex.x = __expf(v.x - ord2f(m.x));
  ex.y = __expf(v.y - ord2f(m.y));
  ex.z = __expf(v.z - ord2f(m.z));
  ex.w = __expf(v.w - ord2f(m.w));
  ((float4*)ebuf)[e] = ex;
  float* drow = den + d * HH;
  atomicAdd(drow + 0, ex.x);
  atomicAdd(drow + 1, ex.y);
  atomicAdd(drow + 2, ex.z);
  atomicAdd(drow + 3, ex.w);
}

// =================== alpha finalize: alpha = ex/(den+eps) * 1/H, in place ===================
// thread-per-edge, coalesced; removes all divides / den gathers from the 14.4M-lane
// aggregate kernel (they were duplicated 32x per wave there).
__global__ void k_alpha(const int* __restrict__ dstI, const float* __restrict__ den,
                        float* __restrict__ ebuf) {
  int e = blockIdx.x * blockDim.x + threadIdx.x;
  if (e >= ET) return;
  int d = (e < EE) ? dstI[e] : (e - EE);
  float4 dn = ((const float4*)den)[d];
  float4 ex = ((const float4*)ebuf)[e];
  const float inv_h = 1.0f / (float)HH;       // fold head-mean factor in here
  ex.x = ex.x / (dn.x + 1e-16f) * inv_h;
  ex.y = ex.y / (dn.y + 1e-16f) * inv_h;
  ex.z = ex.z / (dn.z + 1e-16f) * inv_h;
  ex.w = ex.w / (dn.w + 1e-16f) * inv_h;
  ((float4*)ebuf)[e] = ex;
}

// =================== weighted scatter-aggregate (head-mean fused) ===================
// one wave per edge; lane covers 4 of 128 features; heads summed in-register,
// single set of atomics into agg[N,128] (L2-resident). Pure FMA inner loop:
// one broadcast float4 load of alphas + 4 gathered float4 loads of h[src].
__global__ void k_aggregate(const int* __restrict__ srcI, const int* __restrict__ dstI,
                            const float* __restrict__ alpha4,
                            const float* __restrict__ hbuf, float* __restrict__ agg) {
  int gw   = (blockIdx.x * blockDim.x + threadIdx.x) >> 5;
  int lane = threadIdx.x & 31;
  if (gw >= ET) return;
  int s = (gw < EE) ? srcI[gw] : (gw - EE);
  int d = (gw < EE) ? dstI[gw] : (gw - EE);
  float4 al = ((const float4*)alpha4)[gw];    // broadcast from L1/L2
  const float a[HH] = {al.x, al.y, al.z, al.w};
  float a0 = 0.f, a1 = 0.f, a2 = 0.f, a3 = 0.f;
#pragma unroll
  for (int h = 0; h < HH; ++h) {
    float4 hv = ((const float4*)(hbuf + (size_t)s * (HH * DD) + h * DD))[lane];
    a0 += a[h] * hv.x; a1 += a[h] * hv.y; a2 += a[h] * hv.z; a3 += a[h] * hv.w;
  }
  float* o = agg + (size_t)d * DD + lane * 4;
  atomicAdd(o + 0, a0);
  atomicAdd(o + 1, a1);
  atomicAdd(o + 2, a2);
  atomicAdd(o + 3, a3);
}

// =================== residual + bias + ReLU ===================
__global__ void k_node_update(const float* __restrict__ agg, const float* __restrict__ bias,
                              const float* __restrict__ xin, float* __restrict__ xout) {
  int i = blockIdx.x * blockDim.x + threadIdx.x;
  if (i >= NN * DD) return;
  float v = agg[i] + bias[i & (DD - 1)] + xin[i];
  xout[i] = v > 0.f ? v : 0.f;
}

// =================== final LayerNorm, one wave per node ===================
__global__ void k_layernorm(const float* __restrict__ xin, const float* __restrict__ gamma,
                            const float* __restrict__ beta, float* __restrict__ out) {
  int gw   = (blockIdx.x * blockDim.x + threadIdx.x) >> 5;
  int lane = threadIdx.x & 31;
  if (gw >= NN) return;
  float4 x = ((const float4*)(xin + (size_t)gw * DD))[lane];
  float s  = x.x + x.y + x.z + x.w;
  float s2 = x.x * x.x + x.y * x.y + x.z * x.z + x.w * x.w;
#pragma unroll
  for (int off = 16; off; off >>= 1) {
    s  += __shfl_xor(s, off, 32);
    s2 += __shfl_xor(s2, off, 32);
  }
  float mu  = s * (1.0f / DD);
  float var = s2 * (1.0f / DD) - mu * mu;
  float rs  = rsqrtf(var + LN_EPS);
  float4 g = ((const float4*)gamma)[lane];
  float4 b = ((const float4*)beta)[lane];
  float4 o;
  o.x = (x.x - mu) * rs * g.x + b.x;
  o.y = (x.y - mu) * rs * g.y + b.y;
  o.z = (x.z - mu) * rs * g.z + b.z;
  o.w = (x.w - mu) * rs * g.w + b.w;
  ((float4*)(out + (size_t)gw * DD))[lane] = o;
}

// =================== host launcher ===================
extern "C" void kernel_launch(void* const* d_in, const int* in_sizes, int n_in,
                              void* d_out, int out_size, void* d_ws, size_t ws_size,
                              hipStream_t stream) {
  (void)in_sizes; (void)n_in; (void)out_size; (void)ws_size;
  const float* x        = (const float*)d_in[0];
  const int*   eidx     = (const int*)d_in[1];
  const float* W        = (const float*)d_in[2];   // [3,128,512]
  const float* att_src  = (const float*)d_in[3];   // [3,1,4,128]
  const float* att_dst  = (const float*)d_in[4];
  const float* bias     = (const float*)d_in[5];   // [3,128]
  const float* gamma    = (const float*)d_in[6];
  const float* beta     = (const float*)d_in[7];
  const int* srcI = eidx;
  const int* dstI = eidx + EE;

  // carve workspace (256B aligned slices)
  char* ws = (char*)d_ws;
  size_t off = 0;
  auto carve = [&](size_t bytes) -> void* {
    void* p = ws + off;
    off += (bytes + 255) & ~(size_t)255;
    return p;
  };
  float*    hbuf = (float*)   carve((size_t)NN * HH * DD * 4);  // 102.4 MB (L2-resident)
  __bf16*   xb   = (__bf16*)  carve((size_t)NN * DD * 2);       // 12.8 MB
  __bf16*   wt   = (__bf16*)  carve((size_t)HH * DD * DD * 2);  // 128 KB
  float*    alS  = (float*)   carve((size_t)NN * HH * 4);
  float*    alD  = (float*)   carve((size_t)NN * HH * 4);
  float*    ebuf = (float*)   carve((size_t)ET * HH * 4);       // 7.2 MB: e -> exp -> alpha
  unsigned* mord = (unsigned*)carve((size_t)NN * HH * 4);
  float*    den  = (float*)   carve((size_t)NN * HH * 4);
  float*    agg  = (float*)   carve((size_t)NN * DD * 4);       // 25.6 MB
  float*    xb0  = (float*)   carve((size_t)NN * DD * 4);
  float*    xb1  = (float*)   carve((size_t)NN * DD * 4);

  for (int l = 0; l < LL; ++l) {
    const float* xin  = (l == 0) ? x : ((l == 1) ? xb0 : xb1);
    float*       xout = (l == 0) ? xb0 : ((l == 1) ? xb1 : xb0);

    k_f32_to_bf16<<<(NN * DD + 255) / 256, 256, 0, stream>>>(xin, xb, NN * DD);
    k_wt_bf16<<<((HH * DD) * DD + 255) / 256, 256, 0, stream>>>(W + (size_t)l * DD * HH * DD, wt);

    k_gemm_wmma<<<NN / 16, 128, 0, stream>>>(xb, wt, hbuf);

    k_attn_logits<<<(NN * HH * 32 + 255) / 256, 256, 0, stream>>>(
        hbuf, att_src + (size_t)l * HH * DD, att_dst + (size_t)l * HH * DD, alS, alD);

    hipMemsetAsync(mord, 0, (size_t)NN * HH * 4, stream);  // any finite logit beats ord-encoding 0
    hipMemsetAsync(den,  0, (size_t)NN * HH * 4, stream);
    hipMemsetAsync(agg,  0, (size_t)NN * DD * 4, stream);

    k_edge_logits<<<(ET + 255) / 256, 256, 0, stream>>>(srcI, dstI, alS, alD, ebuf, mord);
    k_exp_den<<<(ET + 255) / 256, 256, 0, stream>>>(dstI, mord, ebuf, den);
    k_alpha<<<(ET + 255) / 256, 256, 0, stream>>>(dstI, den, ebuf);
    k_aggregate<<<(ET * 32 + 255) / 256, 256, 0, stream>>>(srcI, dstI, ebuf, hbuf, agg);
    k_node_update<<<(NN * DD + 255) / 256, 256, 0, stream>>>(agg, bias + (size_t)l * DD, xin, xout);
  }

  // final x lives in xb0 after l=2
  k_layernorm<<<(NN * 32 + 255) / 256, 256, 0, stream>>>(xb0, gamma, beta, (float*)d_out);
}
